// GAT_AE_89240830476552
// MI455X (gfx1250) — matrix operand order repaired and checked by reading, not compiled, not run on hardware
//
#include <hip/hip_runtime.h>
#include <math.h>

// ---------------------------------------------------------------------------
// GAT auto-encoder for MI455X (gfx1250, wave32).
//  * Encoder GEMMs + key projections: v_wmma_f32_16x16x32_f16 (f16 in, f32 acc)
//    with fragment-layout LDS staging (ds_load_b128 / ds_store_b128) and an
//    unguarded fast path + single guarded K-tail tile.
//  * GAT edge softmax: L2 float atomics (max via CAS, add native)
//  * Decoders: 2047-step serial chain, 1 x 1024-thread WG per decoder (2 blocks
//    of one kernel -> concurrent WGPs), state in LDS, k/enc/WoutT streamed from
//    L2 (everything fits in the 192MB L2), float4 dots + prefetch hints.
// ---------------------------------------------------------------------------

typedef __attribute__((ext_vector_type(16))) _Float16 v16h;
typedef __attribute__((ext_vector_type(8)))  _Float16 v8h;
typedef __attribute__((ext_vector_type(8)))  float    v8f;

static constexpr int   Ln    = 2048;
static constexpr int   En    = 32768;
static constexpr int   EA    = En + Ln;       // + self loops
static constexpr int   OD0   = 2001;
static constexpr int   OD1   = 501;

// ---- workspace layout (float offsets, all 64-float aligned) ----------------
static constexpr size_t OFF_XPE  = 0;                               // 2048*2000
static constexpr size_t OFF_H1   = OFF_XPE  + (size_t)Ln*2000;      // 2048*256
static constexpr size_t OFF_AGG1 = OFF_H1   + (size_t)Ln*256;
static constexpr size_t OFF_H2   = OFF_AGG1 + (size_t)Ln*256;       // 2048*64
static constexpr size_t OFF_AGG2 = OFF_H2   + (size_t)Ln*64;
static constexpr size_t OFF_AS1  = OFF_AGG2 + (size_t)Ln*64;        // 2048*4
static constexpr size_t OFF_AD1  = OFF_AS1  + (size_t)Ln*4;
static constexpr size_t OFF_MAX1 = OFF_AD1  + (size_t)Ln*4;
static constexpr size_t OFF_DEN1 = OFF_MAX1 + (size_t)Ln*4;
static constexpr size_t OFF_AS2  = OFF_DEN1 + (size_t)Ln*4;         // 2048
static constexpr size_t OFF_AD2  = OFF_AS2  + (size_t)Ln;
static constexpr size_t OFF_MAX2 = OFF_AD2  + (size_t)Ln;
static constexpr size_t OFF_DEN2 = OFF_MAX2 + (size_t)Ln;
static constexpr size_t OFF_ESC1 = OFF_DEN2 + (size_t)Ln;           // EA*4
static constexpr size_t OFF_ESC2 = OFF_ESC1 + (size_t)EA*4;        // EA
static constexpr size_t OFF_ENC  = OFF_ESC2 + (size_t)EA;          // 4096*64
static constexpr size_t OFF_K0   = OFF_ENC  + (size_t)2*Ln*64;
static constexpr size_t OFF_K1   = OFF_K0   + (size_t)2*Ln*64;
static constexpr size_t OFF_S0   = OFF_K1   + (size_t)2*Ln*64;
static constexpr size_t OFF_S1   = OFF_S0   + 64;
static constexpr size_t OFF_WT0  = OFF_S1   + 64;                   // 2001*192
static constexpr size_t OFF_WT1  = OFF_WT0  + (size_t)OD0*192;      // 501*256

// ---------------------------------------------------------------------------
__device__ inline void atomicMaxF(float* addr, float val) {
  int cur = __float_as_int(*addr);
  while (__int_as_float(cur) < val) {
    int assumed = cur;
    cur = atomicCAS((int*)addr, assumed, __float_as_int(val));
    if (cur == assumed) break;
  }
}

// x*sqrt(d) + positional encoding (module's formula: col j even->sin, odd->cos,
// angle = pos * 10000^(-2j/d))
__global__ void posenc_kernel(const float* __restrict__ x, float* __restrict__ out,
                              int L, int d) {
  size_t i = (size_t)blockIdx.x * blockDim.x + threadIdx.x;
  size_t total = (size_t)L * d;
  if (i >= total) return;
  int pos = (int)(i / d), j = (int)(i % d);
  float ang = (float)pos * powf(10000.f, -2.f * (float)j / (float)d);
  float pe = (j & 1) ? cosf(ang) : sinf(ang);
  out[i] = x[i] * sqrtf((float)d) + pe;
}

// ---- tiled WMMA GEMM: C[M,N] = A[M,K] @ B[K,N] (+bias), f32 io, f16 compute.
// PRECONDITIONS: M % 64 == 0, N % 64 == 0; K arbitrary (tail zero-padded);
// A/B/C bases and row strides 16B-aligned (true for all launches here).
//
// Block 256 threads = 8 waves; block tile 64x64; wave tile 16x32 (2 WMMA accs).
// LDS tiles are kept in *fragment layout*: Asf[waveRowTile][lane] /
// Bsf[colSubTile][lane] hold each lane's 16 halfwords contiguously, so the
// fragment read is a plain v16h load (2x ds_load_b128) and each staging thread
// writes one contiguous 8-half chunk (1x ds_store_b128).
//
// A frag mapping (16x32, ISA): lane = (M%16) + 16*h with h=(c>>3)&1,
//   slot = (c&7) + ((c>>4)<<3).
// B frag mapping (32x16): lane = (n%16) + 16*((kk>>3)&1), slot = (kk&7) + ((kk>>4)<<3) (same as A).
__global__ __launch_bounds__(256)
void gemm_wmma(const float* __restrict__ A, const float* __restrict__ B,
               float* __restrict__ C, const float* __restrict__ bias,
               int M, int N, int K) {
  __shared__ v16h Asf[4][32];   // [wave row tile][lane]   (4 KB)
  __shared__ v16h Bsf[4][32];   // [16-col subtile][lane]  (4 KB)
  const int tid  = threadIdx.x;
  const int wave = tid >> 5, lane = tid & 31;
  const int wm = wave & 3, wn = wave >> 2;        // 4x2 wave grid
  const int bm = blockIdx.y * 64, bn = blockIdx.x * 64;
  const int half = lane >> 4, l16 = lane & 15;

  // A stager: thread -> (row r = tid>>2 in [0,64), c-group acg = tid&3, 8 cols)
  const int ar = tid >> 2, acg = tid & 3;
  const int a_wm    = ar >> 4;
  const int a_lane  = (ar & 15) + 16 * (acg & 1);
  const int a_chunk = acg >> 1;                   // slots 0-7 or 8-15
  // B stager: thread -> (col n = tid&63, k-group bkg = tid>>6, 8 rows)
  const int bcol = tid & 63, bkg = tid >> 6;      // bkg in [0,4)
  const int b_sub   = bcol >> 4;
  const int b_lane  = (bcol & 15) + 16 * (bkg & 1);
  const int b_chunk = bkg >> 1;

  const float* __restrict__ arow = A + (size_t)(bm + ar) * K;
  const float* __restrict__ bcp  = B + bn + bcol;

  v8f acc0 = {}, acc1 = {};
  const int kfull = K & ~31;

  // ---- fast path: full 32-wide K tiles, no bounds checks ----
  for (int k0 = 0; k0 < kfull; k0 += 32) {
    { // stage A: 8 consecutive K elements -> 2x global b128 + 1x ds b128
      const int c0 = k0 + acg * 8;
      float4 f0 = *(const float4*)(arow + c0);
      float4 f1 = *(const float4*)(arow + c0 + 4);
      v8h v;
      v[0] = (_Float16)f0.x; v[1] = (_Float16)f0.y;
      v[2] = (_Float16)f0.z; v[3] = (_Float16)f0.w;
      v[4] = (_Float16)f1.x; v[5] = (_Float16)f1.y;
      v[6] = (_Float16)f1.z; v[7] = (_Float16)f1.w;
      ((v8h*)&Asf[a_wm][a_lane])[a_chunk] = v;
    }
    { // stage B: 8-row column chunk (coalesced across the 64 threads of a kg)
      const size_t r0 = (size_t)(k0 + bkg * 8) * N;
      v8h v;
#pragma unroll
      for (int j = 0; j < 8; ++j) v[j] = (_Float16)bcp[r0 + (size_t)j * N];
      ((v8h*)&Bsf[b_sub][b_lane])[b_chunk] = v;
    }
    __syncthreads();

    v16h afrag = Asf[wm][lane];            // 2x ds_load_b128
    v16h bf0   = Bsf[wn * 2 + 0][lane];
    v16h bf1   = Bsf[wn * 2 + 1][lane];
    acc0 = __builtin_amdgcn_wmma_f32_16x16x32_f16(false, afrag, false, bf0,
                                                  (short)0, acc0, false, false);
    acc1 = __builtin_amdgcn_wmma_f32_16x16x32_f16(false, afrag, false, bf1,
                                                  (short)0, acc1, false, false);
    __syncthreads();
  }

  // ---- guarded tail tile (only when K % 32 != 0) ----
  if (kfull < K) {
    {
      const int c0 = kfull + acg * 8;
      v8h v;
#pragma unroll
      for (int j = 0; j < 8; ++j) {
        int gc = c0 + j;
        v[j] = (_Float16)((gc < K) ? arow[gc] : 0.f);
      }
      ((v8h*)&Asf[a_wm][a_lane])[a_chunk] = v;
    }
    {
      const int r0 = kfull + bkg * 8;
      v8h v;
#pragma unroll
      for (int j = 0; j < 8; ++j) {
        int gr = r0 + j;
        v[j] = (_Float16)((gr < K) ? bcp[(size_t)gr * N] : 0.f);
      }
      ((v8h*)&Bsf[b_sub][b_lane])[b_chunk] = v;
    }
    __syncthreads();

    v16h afrag = Asf[wm][lane];
    v16h bf0   = Bsf[wn * 2 + 0][lane];
    v16h bf1   = Bsf[wn * 2 + 1][lane];
    acc0 = __builtin_amdgcn_wmma_f32_16x16x32_f16(false, afrag, false, bf0,
                                                  (short)0, acc0, false, false);
    acc1 = __builtin_amdgcn_wmma_f32_16x16x32_f16(false, afrag, false, bf1,
                                                  (short)0, acc1, false, false);
  }

  // ---- epilogue (M,N multiples of 64 -> no guards) ----
#pragma unroll
  for (int sub = 0; sub < 2; ++sub) {
    v8f acc = sub ? acc1 : acc0;
    int gc = bn + wn * 32 + sub * 16 + l16;
    float bv = bias ? bias[gc] : 0.f;
#pragma unroll
    for (int g = 0; g < 8; ++g) {
      int gr = bm + wm * 16 + g + 8 * half;       // 16x16 C layout
      C[(size_t)gr * N + gc] = acc[g] + bv;
    }
  }
}

// alpha_{s,d}[n,h] = sum_f h[n,h,f] * a_{src,dst}[h,f]
__global__ void alpha_kernel(const float* __restrict__ h, const float* __restrict__ a_src,
                             const float* __restrict__ a_dst, float* __restrict__ as_,
                             float* __restrict__ ad_, int N, int H, int Fd) {
  int i = blockIdx.x * blockDim.x + threadIdx.x;
  if (i >= N * H) return;
  int hd = i % H;
  const float* hr  = h + (size_t)i * Fd;
  const float* asr = a_src + (size_t)hd * Fd;
  const float* adr = a_dst + (size_t)hd * Fd;
  float s = 0.f, d2 = 0.f;
  for (int f = 0; f < Fd; ++f) { s += hr[f] * asr[f]; d2 += hr[f] * adr[f]; }
  as_[i] = s; ad_[i] = d2;
}

__global__ void fill_kernel(float* __restrict__ p, float v, size_t n) {
  size_t i = (size_t)blockIdx.x * blockDim.x + threadIdx.x;
  if (i < n) p[i] = v;
}

// edge index layout: ei[0..E) = src, ei[E..2E) = dst; e>=E -> self loop (e-E)
__global__ void edge_score_kernel(const int* __restrict__ ei, int E, int N, int H,
                                  const float* __restrict__ as_, const float* __restrict__ ad_,
                                  float* __restrict__ emax, float* __restrict__ esc) {
  int idx = blockIdx.x * blockDim.x + threadIdx.x;
  if (idx >= (E + N) * H) return;
  int e = idx / H, hd = idx - e * H;
  int s = (e < E) ? ei[e] : (e - E);
  int d = (e < E) ? ei[E + e] : (e - E);
  float v = as_[s * H + hd] + ad_[d * H + hd];
  v = (v > 0.f) ? v : 0.2f * v;                   // leaky_relu(0.2)
  esc[idx] = v;
  atomicMaxF(&emax[d * H + hd], v);
}

__global__ void edge_exp_kernel(const int* __restrict__ ei, int E, int N, int H,
                                const float* __restrict__ emax, float* __restrict__ denom,
                                float* __restrict__ esc) {
  int idx = blockIdx.x * blockDim.x + threadIdx.x;
  if (idx >= (E + N) * H) return;
  int e = idx / H, hd = idx - e * H;
  int d = (e < E) ? ei[E + e] : (e - E);
  float v = __expf(esc[idx] - emax[d * H + hd]);
  esc[idx] = v;
  atomicAdd(&denom[d * H + hd], v);
}

__global__ void edge_agg_kernel(const int* __restrict__ ei, int E, int N, int H, int Fd,
                                const float* __restrict__ esc, const float* __restrict__ denom,
                                const float* __restrict__ h, float* __restrict__ agg) {
  size_t idx = (size_t)blockIdx.x * blockDim.x + threadIdx.x;
  size_t total = (size_t)(E + N) * H * Fd;
  if (idx >= total) return;
  int f = (int)(idx % Fd);
  size_t eh = idx / Fd;
  int e = (int)(eh / H), hd = (int)(eh - (size_t)e * H);
  int s = (e < E) ? ei[e] : (e - E);
  int d = (e < E) ? ei[E + e] : (e - E);
  float alpha = esc[eh] / (denom[d * H + hd] + 1e-16f);
  atomicAdd(&agg[((size_t)d * H + hd) * Fd + f],
            h[((size_t)s * H + hd) * Fd + f] * alpha);
}

// out[i] = act(in[i] + b[i % C]); act: 0=none, 1=elu
__global__ void bias_act_kernel(const float* __restrict__ in, const float* __restrict__ b,
                                float* __restrict__ out, int C, size_t n, int act) {
  size_t i = (size_t)blockIdx.x * blockDim.x + threadIdx.x;
  if (i >= n) return;
  float v = in[i] + b[i % C];
  if (act == 1) v = (v > 0.f) ? v : (__expf(v) - 1.f);
  out[i] = v;
}

__global__ void colmean_kernel(const float* __restrict__ m, float* __restrict__ s,
                               int rows, int cols) {
  int c = blockIdx.x * blockDim.x + threadIdx.x;
  if (c >= cols) return;
  float acc = 0.f;
  for (int r = 0; r < rows; ++r) acc += m[(size_t)r * cols + c];
  s[c] = acc / (float)rows;
}

// W [rows,cols] -> WT [cols,rows]
__global__ void transpose_kernel(const float* __restrict__ W, float* __restrict__ WT,
                                 int rows, int cols) {
  size_t i = (size_t)blockIdx.x * blockDim.x + threadIdx.x;
  if (i >= (size_t)rows * cols) return;
  int r = (int)(i / cols), c = (int)(i % cols);
  WT[(size_t)c * rows + r] = W[i];
}

// one-hot row 0 of both outputs
__global__ void initout_kernel(float* __restrict__ out0, float* __restrict__ out1,
                               const int* __restrict__ X0, const int* __restrict__ X1,
                               int od0, int od1) {
  int i = blockIdx.x * blockDim.x + threadIdx.x;
  if (i < od0)            out0[i] = (i == X0[0]) ? 1.f : 0.f;
  else if (i < od0 + od1) { int j = i - od0; out1[j] = (j == X1[0]) ? 1.f : 0.f; }
}

// ---------------------------------------------------------------------------
// Sequential attention-GRU decoder: one 1024-thread block per decoder.
struct DecArgs {
  const float *k, *enc, *sinit, *Wq, *bq, *embA, *embB;
  const int   *tokA, *tokB;
  const float *Wih0, *Whh0, *bih0, *bhh0;
  const float *Wih1, *Whh1, *bih1, *bhh1;
  const float *WoutT, *bout;
  float *out;
  int in0, catdim, odim, taOff, hasB;
};

__device__ inline float sigmoidf_(float x) { return 1.f / (1.f + __expf(-x)); }
__device__ inline float dot4(float4 a, float4 b) {
  return a.x * b.x + a.y * b.y + a.z * b.z + a.w * b.w;
}

__global__ __launch_bounds__(1024)
void decoder_kernel(DecArgs a0, DecArgs a1, int nsteps) {
  DecArgs a = (blockIdx.x == 0) ? a0 : a1;
  __shared__ float sc[4096];
  __shared__ float red[1024];
  __shared__ float svec[64], qv[64], emb1[64], emb2[64], cvec[64], x1v[64], hvec[64];
  __shared__ float inp0[192], giA[192], ghA[192], catv[256];

  const int tid = threadIdx.x;
  if (tid < 64) svec[tid] = a.sinit[tid];
  __syncthreads();

  for (int t = 0; t < nsteps; ++t) {
    // embeddings + q = s@Wq + bq
    if (tid < 64) {
      int ta = a.tokA[t + a.taOff];
      emb1[tid] = a.embA[(size_t)ta * 64 + tid];
      if (a.hasB) { int tb = a.tokB[t]; emb2[tid] = a.embB[(size_t)tb * 64 + tid]; }
      float acc = a.bq[tid];
      for (int i = 0; i < 64; ++i) acc += svec[i] * a.Wq[i * 64 + tid];
      qv[tid] = acc;
    }
    __syncthreads();

    // hoist q into registers (16 float4 = 64 VGPRs; 8 waves/SIMD still fit)
    float4 qr[16];
#pragma unroll
    for (int j = 0; j < 16; ++j) qr[j] = ((const float4*)qv)[j];

    // scores = k@q / sqrt(64)   (k rows are 256B-aligned -> global_load_b128)
    float lm = -3.0e38f;
    for (int n = tid; n < 4096; n += 1024) {
      const float4* kr = (const float4*)(a.k + (size_t)n * 64);
      float acc = 0.f;
#pragma unroll
      for (int j = 0; j < 16; ++j) acc += dot4(kr[j], qr[j]);
      acc *= 0.125f;
      sc[n] = acc;
      lm = fmaxf(lm, acc);
    }
    red[tid] = lm; __syncthreads();
    for (int off = 512; off > 0; off >>= 1) {
      if (tid < off) red[tid] = fmaxf(red[tid], red[tid + off]);
      __syncthreads();
    }
    float gmax = red[0]; __syncthreads();

    float ls = 0.f;
    for (int n = tid; n < 4096; n += 1024) {
      float v = __expf(sc[n] - gmax); sc[n] = v; ls += v;
    }
    red[tid] = ls; __syncthreads();
    for (int off = 512; off > 0; off >>= 1) {
      if (tid < off) red[tid] += red[tid + off];
      __syncthreads();
    }
    float inv = 1.f / red[0]; __syncthreads();

    // c = probs @ enc  (16 partial groups x 64 columns; coalesced over col)
    {
      int col = tid & 63, grp = tid >> 6;
      float acc = 0.f;
      for (int n = grp; n < 4096; n += 16) acc += sc[n] * a.enc[(size_t)n * 64 + col];
      red[tid] = acc; __syncthreads();
      for (int off = 8; off > 0; off >>= 1) {
        if (grp < off) red[tid] += red[tid + off * 64];
        __syncthreads();
      }
      if (grp == 0) cvec[col] = red[col] * inv;
    }
    __syncthreads();

    // gru layer0 input = [emb1 (,emb2), c]
    if (tid < a.in0) {
      float v;
      if (tid < 64) v = emb1[tid];
      else if (tid < a.in0 - 64) v = emb2[tid - 64];
      else v = cvec[tid - (a.in0 - 64)];
      inp0[tid] = v;
    }
    __syncthreads();

    if (tid < 192) {
      float acc = a.bih0[tid];
      const float4* w = (const float4*)(a.Wih0 + (size_t)tid * a.in0);
      for (int i = 0; i < a.in0 / 4; ++i) acc += dot4(w[i], ((const float4*)inp0)[i]);
      giA[tid] = acc;
      float acc2 = a.bhh0[tid];
      const float4* w2 = (const float4*)(a.Whh0 + (size_t)tid * 64);
#pragma unroll
      for (int i = 0; i < 16; ++i) acc2 += dot4(w2[i], ((const float4*)svec)[i]);
      ghA[tid] = acc2;
    }
    __syncthreads();
    if (tid < 64) {
      float r = sigmoidf_(giA[tid] + ghA[tid]);
      float z = sigmoidf_(giA[64 + tid] + ghA[64 + tid]);
      float nn = tanhf(giA[128 + tid] + r * ghA[128 + tid]);
      x1v[tid] = (1.f - z) * nn + z * svec[tid];
    }
    __syncthreads();

    // gru layer1 (input dim 64); hidden for BOTH layers is the carried s
    if (tid < 192) {
      float acc = a.bih1[tid];
      const float4* w = (const float4*)(a.Wih1 + (size_t)tid * 64);
#pragma unroll
      for (int i = 0; i < 16; ++i) acc += dot4(w[i], ((const float4*)x1v)[i]);
      giA[tid] = acc;
      float acc2 = a.bhh1[tid];
      const float4* w2 = (const float4*)(a.Whh1 + (size_t)tid * 64);
#pragma unroll
      for (int i = 0; i < 16; ++i) acc2 += dot4(w2[i], ((const float4*)svec)[i]);
      ghA[tid] = acc2;
    }
    __syncthreads();
    if (tid < 64) {
      float r = sigmoidf_(giA[tid] + ghA[tid]);
      float z = sigmoidf_(giA[64 + tid] + ghA[64 + tid]);
      float nn = tanhf(giA[128 + tid] + r * ghA[128 + tid]);
      hvec[tid] = (1.f - z) * nn + z * svec[tid];
    }
    __syncthreads();

    // pred = [h, c, emb1 (, emb2)] @ Wout + bout, written straight to d_out
    if (tid < a.catdim) {
      float v;
      if (tid < 64) v = hvec[tid];
      else if (tid < 128) v = cvec[tid - 64];
      else if (tid < 192) v = emb1[tid - 128];
      else v = emb2[tid - 192];
      catv[tid] = v;
    }
    __syncthreads();

    for (int o = tid; o < a.odim; o += 1024) {
      const float4* w = (const float4*)(a.WoutT + (size_t)o * a.catdim);
      if (o + 1024 < a.odim)
        __builtin_prefetch(a.WoutT + (size_t)(o + 1024) * a.catdim, 0, 1);
      float acc = a.bout[o];
      for (int i = 0; i < a.catdim / 4; ++i) acc += dot4(w[i], ((const float4*)catv)[i]);
      a.out[(size_t)(t + 1) * a.odim + o] = acc;
    }

    if (tid < 64) svec[tid] = hvec[tid];
    __syncthreads();
  }
}

// ---------------------------------------------------------------------------
extern "C" void kernel_launch(void* const* d_in, const int* in_sizes, int n_in,
                              void* d_out, int out_size, void* d_ws, size_t ws_size,
                              hipStream_t stream) {
  (void)in_sizes; (void)n_in; (void)out_size; (void)ws_size;
  // Input order: x0,x1,edge_index0,edge_index1,X0,X1, then params leaves in
  // dict-insertion order (enc0: W1,as1,ad1,b1,W2,as2,ad2,b2; enc1 same;
  // dec0: Wq,bq,Wk,bk,emb,gru[2x(Wih,Whh,bih,bhh)],Wout,bout; dec1 same with
  // emb_act,emb_attr).
  auto Fp = [&](int i) { return (const float*)d_in[i]; };
  auto Ip = [&](int i) { return (const int*)d_in[i]; };
  float* ws   = (float*)d_ws;
  float* out0 = (float*)d_out;
  float* out1 = out0 + (size_t)Ln * OD0;
  auto cdiv = [](size_t a, size_t b) { return (unsigned)((a + b - 1) / b); };

  auto run_encoder = [&](const float* x, int d, const int* ei, int pb, int rowoff) {
    const float *W1 = Fp(pb+0), *a_s1 = Fp(pb+1), *a_d1 = Fp(pb+2), *b1 = Fp(pb+3);
    const float *W2 = Fp(pb+4), *a_s2 = Fp(pb+5), *a_d2 = Fp(pb+6), *b2 = Fp(pb+7);
    float *xpe = ws+OFF_XPE, *h1 = ws+OFF_H1, *agg1 = ws+OFF_AGG1;
    float *h2 = ws+OFF_H2, *agg2 = ws+OFF_AGG2;
    float *as1 = ws+OFF_AS1, *ad1 = ws+OFF_AD1, *mx1 = ws+OFF_MAX1, *dn1 = ws+OFF_DEN1;
    float *as2 = ws+OFF_AS2, *ad2 = ws+OFF_AD2, *mx2 = ws+OFF_MAX2, *dn2 = ws+OFF_DEN2;
    float *esc1 = ws+OFF_ESC1, *esc2 = ws+OFF_ESC2;

    size_t nx = (size_t)Ln * d;
    posenc_kernel<<<cdiv(nx,256),256,0,stream>>>(x, xpe, Ln, d);
    gemm_wmma<<<dim3(256/64, Ln/64),256,0,stream>>>(xpe, W1, h1, nullptr, Ln, 256, d);

    alpha_kernel<<<cdiv((size_t)Ln*4,256),256,0,stream>>>(h1, a_s1, a_d1, as1, ad1, Ln, 4, 64);
    fill_kernel<<<cdiv((size_t)Ln*4,256),256,0,stream>>>(mx1, -3.0e38f, (size_t)Ln*4);
    fill_kernel<<<cdiv((size_t)Ln*4,256),256,0,stream>>>(dn1, 0.f, (size_t)Ln*4);
    fill_kernel<<<cdiv((size_t)Ln*256,256),256,0,stream>>>(agg1, 0.f, (size_t)Ln*256);
    edge_score_kernel<<<cdiv((size_t)EA*4,256),256,0,stream>>>(ei, En, Ln, 4, as1, ad1, mx1, esc1);
    edge_exp_kernel  <<<cdiv((size_t)EA*4,256),256,0,stream>>>(ei, En, Ln, 4, mx1, dn1, esc1);
    edge_agg_kernel  <<<cdiv((size_t)EA*4*64,256),256,0,stream>>>(ei, En, Ln, 4, 64, esc1, dn1, h1, agg1);
    bias_act_kernel<<<cdiv((size_t)Ln*256,256),256,0,stream>>>(agg1, b1, agg1, 256, (size_t)Ln*256, 1);

    gemm_wmma<<<dim3(1, Ln/64),256,0,stream>>>(agg1, W2, h2, nullptr, Ln, 64, 256);
    alpha_kernel<<<cdiv((size_t)Ln,256),256,0,stream>>>(h2, a_s2, a_d2, as2, ad2, Ln, 1, 64);
    fill_kernel<<<cdiv((size_t)Ln,256),256,0,stream>>>(mx2, -3.0e38f, (size_t)Ln);
    fill_kernel<<<cdiv((size_t)Ln,256),256,0,stream>>>(dn2, 0.f, (size_t)Ln);
    fill_kernel<<<cdiv((size_t)Ln*64,256),256,0,stream>>>(agg2, 0.f, (size_t)Ln*64);
    edge_score_kernel<<<cdiv((size_t)EA,256),256,0,stream>>>(ei, En, Ln, 1, as2, ad2, mx2, esc2);
    edge_exp_kernel  <<<cdiv((size_t)EA,256),256,0,stream>>>(ei, En, Ln, 1, mx2, dn2, esc2);
    edge_agg_kernel  <<<cdiv((size_t)EA*64,256),256,0,stream>>>(ei, En, Ln, 1, 64, esc2, dn2, h2, agg2);
    bias_act_kernel<<<cdiv((size_t)Ln*64,256),256,0,stream>>>(
        agg2, b2, ws + OFF_ENC + (size_t)rowoff * 64, 64, (size_t)Ln*64, 0);
  };

  run_encoder(Fp(0), 2000, Ip(2), 6, 0);     // enc0 -> enc rows [0,2048)
  run_encoder(Fp(1), 500,  Ip(3), 14, Ln);   // enc1 -> enc rows [2048,4096)

  colmean_kernel<<<1,64,0,stream>>>(ws+OFF_ENC,                 ws+OFF_S0, Ln, 64);
  colmean_kernel<<<1,64,0,stream>>>(ws+OFF_ENC+(size_t)Ln*64,   ws+OFF_S1, Ln, 64);

  // k = enc @ Wk + bk for both decoders (4096x64x64 WMMA)
  gemm_wmma<<<dim3(1, 2*Ln/64),256,0,stream>>>(ws+OFF_ENC, Fp(24), ws+OFF_K0, Fp(25), 2*Ln, 64, 64);
  gemm_wmma<<<dim3(1, 2*Ln/64),256,0,stream>>>(ws+OFF_ENC, Fp(39), ws+OFF_K1, Fp(40), 2*Ln, 64, 64);

  transpose_kernel<<<cdiv((size_t)192*OD0,256),256,0,stream>>>(Fp(35), ws+OFF_WT0, 192, OD0);
  transpose_kernel<<<cdiv((size_t)256*OD1,256),256,0,stream>>>(Fp(51), ws+OFF_WT1, 256, OD1);

  initout_kernel<<<cdiv((size_t)(OD0+OD1),256),256,0,stream>>>(out0, out1, Ip(4), Ip(5), OD0, OD1);

  DecArgs a0, a1;
  a0.k = ws+OFF_K0; a0.enc = ws+OFF_ENC; a0.sinit = ws+OFF_S0;
  a0.Wq = Fp(22); a0.bq = Fp(23); a0.embA = Fp(26); a0.embB = nullptr;
  a0.tokA = Ip(4); a0.tokB = nullptr;
  a0.Wih0 = Fp(27); a0.Whh0 = Fp(28); a0.bih0 = Fp(29); a0.bhh0 = Fp(30);
  a0.Wih1 = Fp(31); a0.Whh1 = Fp(32); a0.bih1 = Fp(33); a0.bhh1 = Fp(34);
  a0.WoutT = ws+OFF_WT0; a0.bout = Fp(36); a0.out = out0;
  a0.in0 = 128; a0.catdim = 192; a0.odim = OD0; a0.taOff = 0; a0.hasB = 0;

  a1.k = ws+OFF_K1; a1.enc = ws+OFF_ENC; a1.sinit = ws+OFF_S1;
  a1.Wq = Fp(37); a1.bq = Fp(38); a1.embA = Fp(41); a1.embB = Fp(42);
  a1.tokA = Ip(4); a1.tokB = Ip(5);          // ta = X0[t+1], tr = X1[t]
  a1.Wih0 = Fp(43); a1.Whh0 = Fp(44); a1.bih0 = Fp(45); a1.bhh0 = Fp(46);
  a1.Wih1 = Fp(47); a1.Whh1 = Fp(48); a1.bih1 = Fp(49); a1.bhh1 = Fp(50);
  a1.WoutT = ws+OFF_WT1; a1.bout = Fp(52); a1.out = out1;
  a1.in0 = 192; a1.catdim = 256; a1.odim = OD1; a1.taOff = 1; a1.hasB = 1;

  decoder_kernel<<<2, 1024, 0, stream>>>(a0, a1, Ln - 1);
}
